// nystromFormerHead_64836826300856
// MI455X (gfx1250) — matrix operand order repaired and checked by reading, not compiled, not run on hardware
//
#include <hip/hip_runtime.h>
#include <hip/hip_bf16.h>

// ---------------------------------------------------------------------------
// Types / helpers
// ---------------------------------------------------------------------------
typedef __bf16 bf16_t;
typedef __attribute__((ext_vector_type(16))) __bf16 v16bf;
typedef __attribute__((ext_vector_type(8)))  float  v8f;

__device__ __forceinline__ float bf2f(bf16_t x) {
    unsigned short u = __builtin_bit_cast(unsigned short, x);
    unsigned int v = ((unsigned int)u) << 16;
    return __builtin_bit_cast(float, v);
}
__device__ __forceinline__ bf16_t f2bf(float f) {
    unsigned int u = __builtin_bit_cast(unsigned int, f);
    unsigned int r = u + 0x7FFFu + ((u >> 16) & 1u);   // round-to-nearest-even
    unsigned short h = (unsigned short)(r >> 16);
    return __builtin_bit_cast(bf16_t, h);
}

// Model constants
#define BB     8
#define NVEC   4096
#define DIMM   512
#define HEADS  8
#define DHH    64
#define MLM    256          // landmarks
#define ITERS  6
#define DEPTH  2
#define NCLASS 10
#define KER    33
#define INNER  (HEADS*DHH)  // 512
#define FFD    (4*DIMM)     // 2048
#define BN     (BB*NVEC)    // 32768
#define BH     (BB*HEADS)   // 64

// ---------------------------------------------------------------------------
// Generic strided-batched bf16 WMMA GEMM.  C = scale*(A @ B) + bias (+resid)
// A: [M,K] row-major bf16.  B: transB ? [N,K] : [K,N] row-major bf16.
// Block tile 128x64, K-step 32, 8 waves (256 thr) in a 2(M) x 4(N) wave grid;
// each wave owns a 64x16 strip (4 WMMA accumulators).  LDS double-buffered:
// one s_barrier per K-step, staging of tile kt+1 overlaps WMMA on tile kt.
// Requires: M%128==0, N%64==0, K%32==0 (true for every GEMM in this model).
// ---------------------------------------------------------------------------
#define BLK_M 128
#define BLK_N 64
#define BLK_K 32
#define LDSR  40   // 32 + 8 pad (bf16 elems); 80B row stride keeps 16B align

__global__ __launch_bounds__(256)
void gemm_bf16_wmma(const __hip_bfloat16* __restrict__ A_, long long sA,
                    const __hip_bfloat16* __restrict__ B_, long long sB,
                    const float* __restrict__ bias,
                    float* __restrict__ Cf, long long sCf,
                    __hip_bfloat16* __restrict__ Cb, long long sCb,
                    const float* __restrict__ resid, long long sR,
                    int Mdim, int Ndim, int Kdim,
                    int transB, float scale, int act_gelu)
{
    __shared__ bf16_t As[2][BLK_M][LDSR];
    __shared__ bf16_t Bs[2][BLK_N][LDSR];   // stored [n][k]

    const int batch = blockIdx.z;
    const bf16_t* Ab = (const bf16_t*)A_ + (long long)batch * sA;
    const bf16_t* Bb = (const bf16_t*)B_ + (long long)batch * sB;

    const int tid  = threadIdx.x;
    const int wave = tid >> 5;       // 0..7
    const int lane = tid & 31;
    const int wm   = wave >> 2;      // 0..1 -> M offset wm*64
    const int wn   = wave & 3;       // 0..3 -> N offset wn*16
    const int m0 = blockIdx.y * BLK_M;
    const int n0 = blockIdx.x * BLK_N;

    // stage one 128x32 A tile + 32x64 B tile into LDS buffer `buf`
    auto stage = [&](int kt, int buf) {
        const int k0 = kt * BLK_K;
        // A: 128x32 bf16 = 512 uint4; 2 per thread (coalesced b128)
        #pragma unroll
        for (int u = 0; u < 2; ++u) {
            int c = (tid << 1) + u;        // 0..511
            int r = c >> 2, q = c & 3;     // row, 8-elem quarter
            *(uint4*)(&As[buf][r][q * 8]) =
                *(const uint4*)(Ab + (long long)(m0 + r) * Kdim + k0 + q * 8);
        }
        if (transB) {
            // B is [N,K]: straight copy, 256 uint4, 1 per thread
            int r = tid >> 2, q = tid & 3;
            *(uint4*)(&Bs[buf][r][q * 8]) =
                *(const uint4*)(Bb + (long long)(n0 + r) * Kdim + k0 + q * 8);
        } else {
            // B is [K,N]: coalesced row read, scattered transpose into LDS
            int k  = tid >> 3;             // 0..31
            int ns = (tid & 7) << 3;       // 0..56 step 8
            union { uint4 u4; bf16_t e[8]; } t;
            t.u4 = *(const uint4*)(Bb + (long long)(k0 + k) * Ndim + n0 + ns);
            #pragma unroll
            for (int j = 0; j < 8; ++j) Bs[buf][ns + j][k] = t.e[j];
        }
    };

    v8f acc[4] = {};
    const int nK = Kdim / BLK_K;

    stage(0, 0);
    __syncthreads();
    for (int kt = 0; kt < nK; ++kt) {
        const int cur = kt & 1;
        if (kt + 1 < nK) stage(kt + 1, cur ^ 1);   // overlap with WMMA below

        // ---- B fragment: lane -> n = wn*16 + lane%16; K contiguous --------
        union { uint4 u4[2]; v16bf v; } bfr;
        {
            int n  = (wn << 4) + (lane & 15);
            int ko = (lane < 16) ? 0 : 16;
            bfr.u4[0] = *(const uint4*)(&Bs[cur][n][ko]);
            bfr.u4[1] = *(const uint4*)(&Bs[cur][n][ko + 8]);
        }
        // ---- 4 A fragments / WMMAs ---------------------------------------
        #pragma unroll
        for (int r = 0; r < 4; ++r) {
            union { uint4 u4[2]; v16bf v; } af;
            int m   = (wm << 6) + (r << 4) + (lane & 15);
            int klo = (lane < 16) ? 0 : 16;
            af.u4[0] = *(const uint4*)(&As[cur][m][klo]);
            af.u4[1] = *(const uint4*)(&As[cur][m][klo + 8]);
            acc[r] = __builtin_amdgcn_wmma_f32_16x16x32_bf16(
                false, af.v, false, bfr.v, (short)0, acc[r], false, false);
        }
        __syncthreads();   // single barrier per K-step (double-buffered LDS)
    }

    // ---- epilogue ---------------------------------------------------------
    float*  Cfb = Cf ? Cf + (long long)batch * sCf : nullptr;
    bf16_t* Cbb = Cb ? (bf16_t*)Cb + (long long)batch * sCb : nullptr;
    const float* Rb = resid ? resid + (long long)batch * sR : nullptr;
    const int col = n0 + (wn << 4) + (lane & 15);
    const float bv = bias ? bias[col] : 0.0f;

    #pragma unroll
    for (int r = 0; r < 4; ++r) {
        #pragma unroll
        for (int e = 0; e < 8; ++e) {
            int row = m0 + (wm << 6) + (r << 4) + ((lane < 16) ? e : e + 8);
            float v = acc[r][e] * scale + bv;
            if (act_gelu) {
                float t = tanhf(0.7978845608028654f * (v + 0.044715f * v * v * v));
                v = 0.5f * v * (1.0f + t);
            }
            long long idx = (long long)row * Ndim + col;
            if (Rb)  v += Rb[idx];
            if (Cfb) Cfb[idx] = v;
            if (Cbb) Cbb[idx] = f2bf(v);
        }
    }
}

// ---------------------------------------------------------------------------
// Elementwise / row kernels
// ---------------------------------------------------------------------------
__global__ void lin_kernel(const float* __restrict__ x, const float* __restrict__ w,
                           const float* __restrict__ b, float* __restrict__ h)
{
    long long idx = (long long)blockIdx.x * 256 + threadIdx.x;
    if (idx >= (long long)BN * DIMM) return;
    int d = idx & (DIMM - 1);
    long long bn = idx >> 9;
    h[idx] = x[bn * 2 + 0] * w[d] + x[bn * 2 + 1] * w[DIMM + d] + b[d];
}

__global__ void convert_bf16_kernel(const float* __restrict__ in,
                                    __hip_bfloat16* __restrict__ out, long long n)
{
    long long idx = (long long)blockIdx.x * 256 + threadIdx.x;
    if (idx < n) ((bf16_t*)out)[idx] = f2bf(in[idx]);
}

__global__ __launch_bounds__(128)
void layernorm_kernel(const float* __restrict__ in, const float* __restrict__ g,
                      const float* __restrict__ b, __hip_bfloat16* __restrict__ out)
{
    __shared__ float red[128];
    const long long row = blockIdx.x;
    const float* p = in + row * DIMM;
    const int tid = threadIdx.x;
    float v[4], s = 0.f;
    #pragma unroll
    for (int i = 0; i < 4; ++i) { v[i] = p[tid + i * 128]; s += v[i]; }
    red[tid] = s; __syncthreads();
    for (int st = 64; st > 0; st >>= 1) { if (tid < st) red[tid] += red[tid + st]; __syncthreads(); }
    const float mu = red[0] * (1.0f / DIMM); __syncthreads();
    float vs = 0.f;
    #pragma unroll
    for (int i = 0; i < 4; ++i) { float d = v[i] - mu; vs += d * d; }
    red[tid] = vs; __syncthreads();
    for (int st = 64; st > 0; st >>= 1) { if (tid < st) red[tid] += red[tid + st]; __syncthreads(); }
    const float rstd = rsqrtf(red[0] * (1.0f / DIMM) + 1e-5f);
    bf16_t* o = (bf16_t*)out + row * DIMM;
    #pragma unroll
    for (int i = 0; i < 4; ++i) {
        int c = tid + i * 128;
        o[c] = f2bf((v[i] - mu) * rstd * g[c] + b[c]);
    }
}

__global__ void qkv_split_kernel(const __hip_bfloat16* __restrict__ qkv,
                                 __hip_bfloat16* __restrict__ q,
                                 __hip_bfloat16* __restrict__ k,
                                 __hip_bfloat16* __restrict__ v)
{
    long long idx = (long long)blockIdx.x * 256 + threadIdx.x;   // [b,h,n,d]
    if (idx >= (long long)BH * NVEC * DHH) return;
    int d = idx & 63; long long t = idx >> 6;
    int n = (int)(t & 4095); t >>= 12;
    int hh = (int)(t & 7);  int bb = (int)(t >> 3);
    const bf16_t* src = (const bf16_t*)qkv +
        ((long long)(bb * NVEC + n)) * (3 * INNER) + hh * DHH + d;
    ((bf16_t*)q)[idx] = f2bf(bf2f(src[0]) * 0.125f);   // DH^-0.5
    ((bf16_t*)k)[idx] = src[INNER];
    ((bf16_t*)v)[idx] = src[2 * INNER];
}

__global__ void landmark_mean_kernel(const __hip_bfloat16* __restrict__ in,
                                     __hip_bfloat16* __restrict__ out)
{
    long long idx = (long long)blockIdx.x * 256 + threadIdx.x;   // [bh, M, d]
    if (idx >= (long long)BH * MLM * DHH) return;
    int d = idx & 63;
    int m = (int)((idx >> 6) & (MLM - 1));
    long long bh = idx >> 14;
    const bf16_t* p = (const bf16_t*)in + (bh * NVEC + (long long)m * 16) * DHH + d;
    float s = 0.f;
    #pragma unroll
    for (int j = 0; j < 16; ++j) s += bf2f(p[(long long)j * DHH]);
    ((bf16_t*)out)[idx] = f2bf(s * (1.0f / 16.0f));
}

__global__ __launch_bounds__(256)
void softmax_kernel(__hip_bfloat16* __restrict__ data, int rowlen)
{
    __shared__ float red[256];
    bf16_t* p = (bf16_t*)data + (long long)blockIdx.x * rowlen;
    const int tid = threadIdx.x;
    float mx = -3.0e38f;
    for (int j = tid; j < rowlen; j += 256) mx = fmaxf(mx, bf2f(p[j]));
    red[tid] = mx; __syncthreads();
    for (int s = 128; s > 0; s >>= 1) { if (tid < s) red[tid] = fmaxf(red[tid], red[tid + s]); __syncthreads(); }
    mx = red[0]; __syncthreads();
    float sum = 0.f;
    for (int j = tid; j < rowlen; j += 256) sum += __expf(bf2f(p[j]) - mx);
    red[tid] = sum; __syncthreads();
    for (int s = 128; s > 0; s >>= 1) { if (tid < s) red[tid] += red[tid + s]; __syncthreads(); }
    const float inv = 1.0f / red[0];
    for (int j = tid; j < rowlen; j += 256) p[j] = f2bf(__expf(bf2f(p[j]) - mx) * inv);
}

// z0 = x^T / (max_rowsum(|x|) * max_colsum(|x|)) per 256x256 batch matrix
__global__ __launch_bounds__(256)
void ns_init_kernel(const __hip_bfloat16* __restrict__ x_, __hip_bfloat16* __restrict__ z_)
{
    __shared__ float cs[256], rs[256];
    const bf16_t* x = (const bf16_t*)x_ + (long long)blockIdx.x * MLM * MLM;
    bf16_t* z = (bf16_t*)z_ + (long long)blockIdx.x * MLM * MLM;
    const int j = threadIdx.x;
    float c = 0.f, r = 0.f;
    for (int i = 0; i < MLM; ++i) {
        c += fabsf(bf2f(x[i * MLM + j]));
        r += fabsf(bf2f(x[j * MLM + i]));
    }
    cs[j] = c; rs[j] = r; __syncthreads();
    for (int s = 128; s > 0; s >>= 1) {
        if (j < s) { cs[j] = fmaxf(cs[j], cs[j + s]); rs[j] = fmaxf(rs[j], rs[j + s]); }
        __syncthreads();
    }
    const float inv = 1.0f / (cs[0] * rs[0]);
    for (int i = 0; i < MLM; ++i) z[j * MLM + i] = f2bf(bf2f(x[i * MLM + j]) * inv);
}

// T = alpha*I - U   (batched 256x256 bf16)
__global__ void diag_affine_kernel(const __hip_bfloat16* __restrict__ U,
                                   __hip_bfloat16* __restrict__ T, float alpha, long long n)
{
    long long idx = (long long)blockIdx.x * 256 + threadIdx.x;
    if (idx >= n) return;
    int j = (int)(idx & (MLM - 1));
    int i = (int)((idx >> 8) & (MLM - 1));
    float v = ((i == j) ? alpha : 0.0f) - bf2f(((const bf16_t*)U)[idx]);
    ((bf16_t*)T)[idx] = f2bf(v);
}

// merged[b,n,h*64+d] = attn_acc[b,h,n,d] + depthwise_conv33(v)[b,h,n,d]
__global__ void attn_merge_kernel(const float* __restrict__ acc,
                                  const __hip_bfloat16* __restrict__ v,
                                  const float* __restrict__ convw,
                                  __hip_bfloat16* __restrict__ merged)
{
    long long idx = (long long)blockIdx.x * 256 + threadIdx.x;   // [bh, n, d]
    if (idx >= (long long)BH * NVEC * DHH) return;
    int d = idx & 63;
    int n = (int)((idx >> 6) & (NVEC - 1));
    long long bh = idx >> 18;
    int hh = (int)(bh & 7), bb = (int)(bh >> 3);
    float val = acc[idx];
    const float* w = convw + hh * KER;
    #pragma unroll
    for (int t = 0; t < KER; ++t) {
        int nn = n + t - KER / 2;
        if (nn >= 0 && nn < NVEC)
            val += w[t] * bf2f(((const bf16_t*)v)[(bh * NVEC + nn) * DHH + d]);
    }
    ((bf16_t*)merged)[((long long)(bb * NVEC + n)) * INNER + hh * DHH + d] = f2bf(val);
}

__global__ __launch_bounds__(256)
void final_head_kernel(const float* __restrict__ h, const float* __restrict__ W,
                       const float* __restrict__ bias, float* __restrict__ out)
{
    __shared__ float red[256];
    const int bb = blockIdx.x / NCLASS, c = blockIdx.x % NCLASS;
    const long long K = (long long)NVEC * DIMM;
    const float* hp = h + (long long)bb * K;
    float s = 0.f;
    for (long long k = threadIdx.x; k < K; k += 256) s += hp[k] * W[k * NCLASS + c];
    red[threadIdx.x] = s; __syncthreads();
    for (int st = 128; st > 0; st >>= 1) { if (threadIdx.x < st) red[threadIdx.x] += red[threadIdx.x + st]; __syncthreads(); }
    if (threadIdx.x == 0) out[blockIdx.x] = red[0] + bias[c];
}

// ---------------------------------------------------------------------------
// Host orchestration
// ---------------------------------------------------------------------------
static inline void gemm(hipStream_t st,
                        const void* A, long long sA, const void* Bp, long long sB,
                        const float* bias, float* Cf, long long sCf,
                        void* Cb, long long sCb, const float* R, long long sR,
                        int M, int N, int K, int batch, int transB,
                        float scale, int gelu)
{
    dim3 g(N / BLK_N, M / BLK_M, batch), blk(256);
    gemm_bf16_wmma<<<g, blk, 0, st>>>(
        (const __hip_bfloat16*)A, sA, (const __hip_bfloat16*)Bp, sB,
        bias, Cf, sCf, (__hip_bfloat16*)Cb, sCb, R, sR,
        M, N, K, transB, scale, gelu);
}

#define EWGRID(n) dim3((unsigned)(((n) + 255) / 256)), dim3(256), 0, stream

extern "C" void kernel_launch(void* const* d_in, const int* in_sizes, int n_in,
                              void* d_out, int out_size, void* d_ws, size_t ws_size,
                              hipStream_t stream)
{
    (void)in_sizes; (void)n_in; (void)out_size; (void)ws_size;
    const float* x       = (const float*)d_in[0];
    const float* lin_w   = (const float*)d_in[1];
    const float* lin_b   = (const float*)d_in[2];
    const float* ln1_g   = (const float*)d_in[3];
    const float* ln1_b   = (const float*)d_in[4];
    const float* qkv_w   = (const float*)d_in[5];
    const float* out_w   = (const float*)d_in[6];
    const float* out_b   = (const float*)d_in[7];
    const float* conv_w  = (const float*)d_in[8];
    const float* ln2_g   = (const float*)d_in[9];
    const float* ln2_b   = (const float*)d_in[10];
    const float* ff1_w   = (const float*)d_in[11];
    const float* ff1_b   = (const float*)d_in[12];
    const float* ff2_w   = (const float*)d_in[13];
    const float* ff2_b   = (const float*)d_in[14];
    const float* final_w = (const float*)d_in[15];
    const float* final_b = (const float*)d_in[16];
    float* out = (float*)d_out;

    // ---- workspace layout -------------------------------------------------
    char* base = (char*)d_ws; size_t off = 0;
    auto alloc = [&](size_t bytes) -> void* {
        off = (off + 255) & ~(size_t)255;
        void* p = base + off; off += bytes; return p;
    };
    const long long QN = (long long)BH * NVEC * DHH;     // 16.78M
    float*          h      = (float*)         alloc((size_t)BN * DIMM * 4);
    __hip_bfloat16* lnbuf  = (__hip_bfloat16*)alloc((size_t)BN * DIMM * 2);      // ln1 / merged / ln2
    __hip_bfloat16* qkvb   = (__hip_bfloat16*)alloc((size_t)BN * 3 * INNER * 2);
    __hip_bfloat16* qb     = (__hip_bfloat16*)alloc((size_t)QN * 2);
    __hip_bfloat16* kb     = (__hip_bfloat16*)alloc((size_t)QN * 2);
    __hip_bfloat16* vb     = (__hip_bfloat16*)alloc((size_t)QN * 2);
    __hip_bfloat16* qlb    = (__hip_bfloat16*)alloc((size_t)BH * MLM * DHH * 2);
    __hip_bfloat16* klb    = (__hip_bfloat16*)alloc((size_t)BH * MLM * DHH * 2);
    __hip_bfloat16* attn1  = (__hip_bfloat16*)alloc((size_t)BH * NVEC * MLM * 2);
    __hip_bfloat16* attn3  = (__hip_bfloat16*)alloc((size_t)BH * MLM * NVEC * 2);
    __hip_bfloat16* big0   = (__hip_bfloat16*)alloc((size_t)BN * FFD * 2);       // az / ffn hidden
    __hip_bfloat16* a2     = (__hip_bfloat16*)alloc((size_t)BH * MLM * MLM * 2);
    __hip_bfloat16* zz     = (__hip_bfloat16*)alloc((size_t)BH * MLM * MLM * 2);
    __hip_bfloat16* z2     = (__hip_bfloat16*)alloc((size_t)BH * MLM * MLM * 2);
    __hip_bfloat16* xz     = (__hip_bfloat16*)alloc((size_t)BH * MLM * MLM * 2);
    __hip_bfloat16* tb     = (__hip_bfloat16*)alloc((size_t)BH * MLM * MLM * 2);
    __hip_bfloat16* ub     = (__hip_bfloat16*)alloc((size_t)BH * MLM * MLM * 2);
    __hip_bfloat16* kvagg  = (__hip_bfloat16*)alloc((size_t)BH * MLM * DHH * 2);
    float*          a_acc  = (float*)         alloc((size_t)QN * 4);
    __hip_bfloat16* wbuf   = (__hip_bfloat16*)alloc((size_t)DIMM * FFD * 2);     // largest weight

    const long long MM64 = (long long)BH * MLM * MLM;

    // ---- input projection -------------------------------------------------
    lin_kernel<<<EWGRID((long long)BN * DIMM)>>>(x, lin_w, lin_b, h);

    for (int i = 0; i < DEPTH; ++i) {
        // ------- attention branch -------
        layernorm_kernel<<<dim3(BN), dim3(128), 0, stream>>>(
            h, ln1_g + i * DIMM, ln1_b + i * DIMM, lnbuf);
        convert_bf16_kernel<<<EWGRID((long long)DIMM * 3 * INNER)>>>(
            qkv_w + (long long)i * DIMM * 3 * INNER, wbuf, (long long)DIMM * 3 * INNER);
        gemm(stream, lnbuf, 0, wbuf, 0, nullptr, nullptr, 0, qkvb, 0, nullptr, 0,
             BN, 3 * INNER, DIMM, 1, 0, 1.0f, 0);
        qkv_split_kernel<<<EWGRID(QN)>>>(qkvb, qb, kb, vb);
        landmark_mean_kernel<<<EWGRID((long long)BH * MLM * DHH)>>>(qb, qlb);
        landmark_mean_kernel<<<EWGRID((long long)BH * MLM * DHH)>>>(kb, klb);

        // attn1 = softmax(q @ k_l^T)  [bh, 4096, 256]
        gemm(stream, qb, (long long)NVEC * DHH, klb, (long long)MLM * DHH,
             nullptr, nullptr, 0, attn1, (long long)NVEC * MLM, nullptr, 0,
             NVEC, MLM, DHH, BH, 1, 1.0f, 0);
        softmax_kernel<<<dim3(BH * NVEC), dim3(256), 0, stream>>>(attn1, MLM);
        // attn2 = softmax(q_l @ k_l^T)  [bh, 256, 256]
        gemm(stream, qlb, (long long)MLM * DHH, klb, (long long)MLM * DHH,
             nullptr, nullptr, 0, a2, (long long)MLM * MLM, nullptr, 0,
             MLM, MLM, DHH, BH, 1, 1.0f, 0);
        softmax_kernel<<<dim3(BH * MLM), dim3(256), 0, stream>>>(a2, MLM);
        // attn3 = softmax(q_l @ k^T)  [bh, 256, 4096]
        gemm(stream, qlb, (long long)MLM * DHH, kb, (long long)NVEC * DHH,
             nullptr, nullptr, 0, attn3, (long long)MLM * NVEC, nullptr, 0,
             MLM, NVEC, DHH, BH, 1, 1.0f, 0);
        softmax_kernel<<<dim3(BH * MLM), dim3(256), 0, stream>>>(attn3, NVEC);

        // ------- Moore-Penrose pinv via Newton-Schulz (batched WMMA) -------
        ns_init_kernel<<<dim3(BH), dim3(256), 0, stream>>>(a2, zz);
        __hip_bfloat16 *zc = zz, *zn = z2;
        const long long sMM = (long long)MLM * MLM;
        for (int it = 0; it < ITERS; ++it) {
            gemm(stream, a2, sMM, zc, sMM, nullptr, nullptr, 0, xz, sMM, nullptr, 0,
                 MLM, MLM, MLM, BH, 0, 1.0f, 0);
            diag_affine_kernel<<<EWGRID(MM64)>>>(xz, tb, 7.0f, MM64);
            gemm(stream, xz, sMM, tb, sMM, nullptr, nullptr, 0, ub, sMM, nullptr, 0,
                 MLM, MLM, MLM, BH, 0, 1.0f, 0);
            diag_affine_kernel<<<EWGRID(MM64)>>>(ub, tb, 15.0f, MM64);
            gemm(stream, xz, sMM, tb, sMM, nullptr, nullptr, 0, ub, sMM, nullptr, 0,
                 MLM, MLM, MLM, BH, 0, 1.0f, 0);
            diag_affine_kernel<<<EWGRID(MM64)>>>(ub, tb, 13.0f, MM64);
            gemm(stream, zc, sMM, tb, sMM, nullptr, nullptr, 0, zn, sMM, nullptr, 0,
                 MLM, MLM, MLM, BH, 0, 0.25f, 0);
            __hip_bfloat16* t = zc; zc = zn; zn = t;
        }

        // kvagg = attn3 @ v   [bh, 256, 64]
        gemm(stream, attn3, (long long)MLM * NVEC, vb, (long long)NVEC * DHH,
             nullptr, nullptr, 0, kvagg, (long long)MLM * DHH, nullptr, 0,
             MLM, DHH, NVEC, BH, 0, 1.0f, 0);
        // az = attn1 @ pinv   [bh, 4096, 256]
        gemm(stream, attn1, (long long)NVEC * MLM, zc, sMM,
             nullptr, nullptr, 0, big0, (long long)NVEC * MLM, nullptr, 0,
             NVEC, MLM, MLM, BH, 0, 1.0f, 0);
        // attn_acc = az @ kvagg   [bh, 4096, 64] fp32
        gemm(stream, big0, (long long)NVEC * MLM, kvagg, (long long)MLM * DHH,
             nullptr, a_acc, (long long)NVEC * DHH, nullptr, 0, nullptr, 0,
             NVEC, DHH, MLM, BH, 0, 1.0f, 0);
        // + depthwise conv residual, merge heads -> [BN, INNER] bf16
        attn_merge_kernel<<<EWGRID(QN)>>>(a_acc, vb, conv_w + (long long)i * HEADS * KER, lnbuf);
        // out-proj + bias + residual into h
        convert_bf16_kernel<<<EWGRID((long long)INNER * DIMM)>>>(
            out_w + (long long)i * INNER * DIMM, wbuf, (long long)INNER * DIMM);
        gemm(stream, lnbuf, 0, wbuf, 0, out_b + i * DIMM, h, 0, nullptr, 0, h, 0,
             BN, DIMM, INNER, 1, 0, 1.0f, 0);

        // ------- FFN branch -------
        layernorm_kernel<<<dim3(BN), dim3(128), 0, stream>>>(
            h, ln2_g + i * DIMM, ln2_b + i * DIMM, lnbuf);
        convert_bf16_kernel<<<EWGRID((long long)DIMM * FFD)>>>(
            ff1_w + (long long)i * DIMM * FFD, wbuf, (long long)DIMM * FFD);
        gemm(stream, lnbuf, 0, wbuf, 0, ff1_b + i * FFD, nullptr, 0, big0, 0, nullptr, 0,
             BN, FFD, DIMM, 1, 0, 1.0f, 1 /*gelu*/);
        convert_bf16_kernel<<<EWGRID((long long)FFD * DIMM)>>>(
            ff2_w + (long long)i * FFD * DIMM, wbuf, (long long)FFD * DIMM);
        gemm(stream, big0, 0, wbuf, 0, ff2_b + i * DIMM, h, 0, nullptr, 0, h, 0,
             BN, DIMM, FFD, 1, 0, 1.0f, 0);
    }

    // ---- classifier head ----------------------------------------------------
    final_head_kernel<<<dim3(BB * NCLASS), dim3(256), 0, stream>>>(h, final_w, final_b, out);
}
